// RNNModel_78795470012954
// MI455X (gfx1250) — matrix operand order repaired and checked by reading, not compiled, not run on hardware
//
#include <hip/hip_runtime.h>
#include <hip/hip_bf16.h>

typedef __attribute__((ext_vector_type(16))) __bf16 v16bf;
typedef __attribute__((ext_vector_type(8)))  __bf16 v8bf;
typedef __attribute__((ext_vector_type(8)))  float  v8f;

#define WMMA_BF16(a, b, c) \
  __builtin_amdgcn_wmma_f32_16x16x32_bf16(false, (a), false, (b), (short)0, (c), false, false)

constexpr int B_    = 32;
constexpr int T_    = 512;
constexpr int Z_    = 128;
constexpr int AE_   = 16;
constexpr int H_    = 1024;
constexpr int G4    = 4 * H_;      // 4096 gate width
constexpr int F_    = Z_ + AE_;    // 144
constexpr int FP    = 160;         // 144 padded up to multiple of 32
constexpr int OUT_  = 1921;
constexpr int OUTP  = 1936;        // 121 * 16
constexpr int TM1   = T_ - 1;      // 511
constexpr int ROWS1 = B_ * T_;     // 16384
constexpr int ROWS3 = B_ * TM1;    // 16352
constexpr int KMIX  = 640;
constexpr int NWG   = 32;          // workgroups cooperating on the recurrence
constexpr int LWAVES = 2;          // waves per recurrence workgroup (64 owner-waves)
constexpr int JWG   = 32;          // hidden units owned per workgroup

// ---------------------------------------------------------------------------
// WMMA fragment loaders (wave32, 16x16x32 bf16)
// A (16x32, row-major src): lanes m<16 hold K 0..15, lanes 16..31 hold K 16..31 (contiguous 16)
// B (32x16, element (k,n) at col_base[n*ldk + k]):
//   lanes n<16 hold K 0..15, lanes 16..31 hold K 16..31 (contiguous 16)
// ---------------------------------------------------------------------------
__device__ __forceinline__ v16bf load_a_frag(const __bf16* row_base_kk, int lane) {
  const int kb = (lane < 16) ? 0 : 16;
  return *(const v16bf*)(row_base_kk + kb);
}

__device__ __forceinline__ v16bf load_b_frag(const __bf16* col_base_kk, int lane) {
  const int kb = (lane < 16) ? 0 : 16;
  return *(const v16bf*)(col_base_kk + kb);
}

// ---------------------------------------------------------------------------
// Kernel 0: pack inputs/weights into padded bf16, fuse biases, zero h/ctr
// ---------------------------------------------------------------------------
__global__ void prep_kernel(const float* __restrict__ z, const int* __restrict__ actions,
                            const float* __restrict__ embed,
                            const float* __restrict__ W_ih, const float* __restrict__ W_hh,
                            const float* __restrict__ b_ih, const float* __restrict__ b_hh,
                            const float* __restrict__ W_out,
                            __bf16* __restrict__ inp_bf, __bf16* __restrict__ Wih_bf,
                            __bf16* __restrict__ Whh_bf, __bf16* __restrict__ Wout_bf,
                            float* __restrict__ bias,
                            __bf16* __restrict__ h0, __bf16* __restrict__ h1,
                            unsigned int* __restrict__ ctr) {
  const long tid = (long)blockIdx.x * blockDim.x + threadIdx.x;
  const long stride = (long)gridDim.x * blockDim.x;
  if (tid == 0) *ctr = 0u;
  for (long i = tid; i < (long)B_ * H_; i += stride) {
    h0[i] = (__bf16)0.f;
    h1[i] = (__bf16)0.f;
  }
  for (long i = tid; i < (long)ROWS1 * FP; i += stride) {
    int r = (int)(i / FP), c = (int)(i % FP);
    float v = 0.f;
    if (c < Z_)      v = z[(long)r * Z_ + c];
    else if (c < F_) v = embed[actions[r] * AE_ + (c - Z_)];
    inp_bf[i] = (__bf16)v;
  }
  for (long i = tid; i < (long)G4 * FP; i += stride) {
    int g = (int)(i / FP), k = (int)(i % FP);
    Wih_bf[i] = (__bf16)((k < F_) ? W_ih[(long)g * F_ + k] : 0.f);
  }
  for (long i = tid; i < (long)G4 * H_; i += stride) Whh_bf[i] = (__bf16)W_hh[i];
  for (long i = tid; i < (long)OUTP * H_; i += stride) {
    int nrow = (int)(i / H_);
    Wout_bf[i] = (__bf16)((nrow < OUT_) ? W_out[i] : 0.f);
  }
  for (long i = tid; i < G4; i += stride) bias[i] = b_ih[i] + b_hh[i];
}

// ---------------------------------------------------------------------------
// Kernel 1: x_pre = inp @ W_ih^T + (b_ih + b_hh)   [16384 x 4096]
// 256 threads = 8 waves, one 16x16 tile per wave
// ---------------------------------------------------------------------------
__global__ void xpre_kernel(const __bf16* __restrict__ inp, const __bf16* __restrict__ Wih,
                            const float* __restrict__ bias, float* __restrict__ x_pre) {
  const int lane = threadIdx.x & 31;
  const int wave = threadIdx.x >> 5;
  const int row0 = blockIdx.y * 16;
  const int col0 = (blockIdx.x * 8 + wave) * 16;
  const __bf16* arow = inp + (long)(row0 + (lane & 15)) * FP;
  const __bf16* bcol = Wih + (long)(col0 + (lane & 15)) * FP;
  v8f acc = {};
#pragma unroll
  for (int kk = 0; kk < FP; kk += 32) {
    v16bf a = load_a_frag(arow + kk, lane);
    v16bf b = load_b_frag(bcol + kk, lane);
    acc = WMMA_BF16(a, b, acc);
  }
  const int n = lane & 15;
  const int mbase = (lane < 16) ? 0 : 8;
#pragma unroll
  for (int r = 0; r < 8; ++r) {
    const int m = row0 + mbase + r;
    const int c = col0 + n;
    x_pre[(long)m * G4 + c] = acc[r] + bias[c];
  }
}

// ---------------------------------------------------------------------------
// Kernel 2: LSTM recurrence across NWG=32 cooperating workgroups (lockstep).
// Each WG owns 32 hidden units; its W_hh slab (128 gate-cols x 1024 = 256 KB)
// is staged into LDS ONCE and reused for all 511 steps. The per-step h
// (64 KB) is staged into LDS after each barrier, so the inner k-loop is
// entirely LDS-fed. 256 + 64 = 320 KB = full WGP LDS. c stays in registers;
// h ping-pongs through two global bf16 buffers; device-wide lockstep via a
// monotonically increasing atomic arrive counter (+ s_sleep spin).
// ---------------------------------------------------------------------------
__global__ void __launch_bounds__(32 * LWAVES, 1)
lstm_kernel(const float* __restrict__ x_pre, const int* __restrict__ dones,
            const __bf16* __restrict__ Whh, __bf16* __restrict__ hbuf0,
            __bf16* __restrict__ hbuf1, __bf16* __restrict__ hs,
            unsigned int* __restrict__ ctr) {
  __shared__ __align__(32) __bf16 sW[4 * JWG * H_];  // 256 KB: [g*32+jj][k]
  __shared__ __align__(32) __bf16 sH[B_ * H_];       // 64 KB:  [m][k]

  const int lane = threadIdx.x & 31;
  const int wave = threadIdx.x >> 5;
  const int bx = blockIdx.x;                  // 0..31
  const int j0 = bx * JWG + wave * 16;        // global hidden slab base
  const int jl0 = wave * 16;                  // local (in-WG) slab base
  const int n = lane & 15;
  const int mbase = (lane < 16) ? 0 : 8;

  // ---- one-time stage of this WG's W_hh slab into LDS ----
  // local col c in [0,128): gate g = c/32, unit jj = c%32
  // global col = g*1024 + bx*32 + jj ; rows of 1024 bf16, 8-elem chunks
  for (int ch = threadIdx.x; ch < (4 * JWG * H_) / 8; ch += 32 * LWAVES) {
    const int col = ch >> 7;                 // 128 chunks per 1024-row
    const int pos = (ch & 127) << 3;
    const int g = col >> 5, jj = col & 31;
    const v8bf v = *(const v8bf*)(Whh + ((long)(g * H_ + bx * JWG + jj) << 10) + pos);
    *(v8bf*)(sW + ((long)col << 10) + pos) = v;
  }
  __syncthreads();

  v8f creg[2];
  creg[0] = (v8f){0.f, 0.f, 0.f, 0.f, 0.f, 0.f, 0.f, 0.f};
  creg[1] = (v8f){0.f, 0.f, 0.f, 0.f, 0.f, 0.f, 0.f, 0.f};
  __bf16* bufs[2] = {hbuf0, hbuf1};

#pragma unroll 1
  for (int t = 0; t < TM1; ++t) {
    const __bf16* hcur = bufs[t & 1];
    __bf16* hnext = bufs[(t + 1) & 1];

    // ---- stage current h (all 32 batches x 1024) into LDS ----
    for (int ch = threadIdx.x; ch < (B_ * H_) / 8; ch += 32 * LWAVES) {
      const int pos = ch << 3;
      *(v8bf*)(sH + pos) = *(const v8bf*)(hcur + pos);
    }
    __syncthreads();

    v8f acc[4][2];
#pragma unroll
    for (int g = 0; g < 4; ++g)
#pragma unroll
      for (int mi = 0; mi < 2; ++mi)
        acc[g][mi] = (v8f){0.f, 0.f, 0.f, 0.f, 0.f, 0.f, 0.f, 0.f};

#pragma unroll 2
    for (int kk = 0; kk < H_; kk += 32) {
      v16bf af[2];
#pragma unroll
      for (int mi = 0; mi < 2; ++mi)
        af[mi] = load_a_frag(sH + (long)(mi * 16 + n) * H_ + kk, lane);
#pragma unroll
      for (int g = 0; g < 4; ++g) {
        v16bf bf_ = load_b_frag(sW + (long)(g * JWG + jl0 + n) * H_ + kk, lane);
#pragma unroll
        for (int mi = 0; mi < 2; ++mi)
          acc[g][mi] = WMMA_BF16(af[mi], bf_, acc[g][mi]);
      }
    }

    // ---- elementwise LSTM cell (c in registers, same tile layout) ----
#pragma unroll
    for (int mi = 0; mi < 2; ++mi) {
      const int m0 = mi * 16 + mbase;
      if (t + 1 < TM1)  // prefetch next step's gate rows (global_prefetch_b8)
        __builtin_prefetch(&x_pre[((long)m0 * T_ + t + 1) * G4 + j0 + n], 0, 1);
#pragma unroll
      for (int r = 0; r < 8; ++r) {
        const int m = m0 + r;            // batch index
        const int j = j0 + n;            // hidden unit
        const long xb = ((long)m * T_ + t) * G4;
        const float gi = acc[0][mi][r] + x_pre[xb + 0 * H_ + j];
        const float gf = acc[1][mi][r] + x_pre[xb + 1 * H_ + j];
        const float gg = acc[2][mi][r] + x_pre[xb + 2 * H_ + j];
        const float go = acc[3][mi][r] + x_pre[xb + 3 * H_ + j];
        const float si = 1.f / (1.f + __expf(-gi));
        const float sf = 1.f / (1.f + __expf(-gf));
        const float so = 1.f / (1.f + __expf(-go));
        const float c_new = sf * creg[mi][r] + si * tanhf(gg);
        const float h_new = so * tanhf(c_new);
        const float mask = 1.f - (float)dones[m * T_ + t];
        creg[mi][r] = c_new * mask;
        hnext[m * H_ + j] = (__bf16)(h_new * mask);
        hs[((long)m * TM1 + t) * H_ + j] = (__bf16)h_new;  // unmasked output
      }
    }

    // ---- device-wide lockstep barrier ----
    __threadfence();     // release hnext stores at device scope
    __syncthreads();     // all waves in this WG arrived (also guards sH reuse)
    if (threadIdx.x == 0) {
      __hip_atomic_fetch_add(ctr, 1u, __ATOMIC_ACQ_REL, __HIP_MEMORY_SCOPE_AGENT);
      const unsigned int target = (unsigned int)(t + 1) * (unsigned int)NWG;
      while (__hip_atomic_load(ctr, __ATOMIC_ACQUIRE, __HIP_MEMORY_SCOPE_AGENT) < target)
        __builtin_amdgcn_s_sleep(2);
    }
    __syncthreads();     // broadcast completion (acquire-inv done by thread 0)
  }
}

// ---------------------------------------------------------------------------
// Kernel 3: out = hs @ W_out^T + b_out, scattered into the 4 output regions
// ---------------------------------------------------------------------------
__global__ void out_kernel(const __bf16* __restrict__ hs, const __bf16* __restrict__ Wout,
                           const float* __restrict__ b_out,
                           float* __restrict__ o_logmix, float* __restrict__ o_mu,
                           float* __restrict__ o_logstd, float* __restrict__ o_done) {
  const int lane = threadIdx.x & 31;
  const int wave = threadIdx.x >> 5;
  const int tn = blockIdx.x * 8 + wave;
  if (tn >= OUTP / 16) return;  // whole-wave early exit -> EXEC all ones for WMMA
  const int row0 = blockIdx.y * 16;
  const int col0 = tn * 16;
  const __bf16* arow = hs + (long)(row0 + (lane & 15)) * H_;
  const __bf16* bcol = Wout + (long)(col0 + (lane & 15)) * H_;
  v8f acc = {};
#pragma unroll 4
  for (int kk = 0; kk < H_; kk += 32) {
    v16bf a = load_a_frag(arow + kk, lane);
    v16bf b = load_b_frag(bcol + kk, lane);
    acc = WMMA_BF16(a, b, acc);
  }
  const int nn = lane & 15;
  const int mbase = (lane < 16) ? 0 : 8;
#pragma unroll
  for (int r = 0; r < 8; ++r) {
    const int row = row0 + mbase + r;
    const int col = col0 + nn;
    if (col >= OUT_) continue;
    const float v = acc[r] + b_out[col];
    if (col < KMIX)          o_logmix[(long)row * KMIX + col] = v;
    else if (col < 2 * KMIX) o_mu[(long)row * KMIX + (col - KMIX)] = v;
    else if (col < 3 * KMIX) o_logstd[(long)row * KMIX + (col - 2 * KMIX)] = v;
    else                     o_done[row] = v;
  }
}

// ---------------------------------------------------------------------------
// Kernel 4: in-place logsumexp over groups of 5 mixture logits
// ---------------------------------------------------------------------------
__global__ void lse_kernel(float* __restrict__ logmix, long ngroups) {
  const long g = (long)blockIdx.x * blockDim.x + threadIdx.x;
  if (g >= ngroups) return;
  float* p = logmix + g * 5;
  const float v0 = p[0], v1 = p[1], v2 = p[2], v3 = p[3], v4 = p[4];
  const float mx = fmaxf(fmaxf(fmaxf(v0, v1), fmaxf(v2, v3)), v4);
  const float s = __expf(v0 - mx) + __expf(v1 - mx) + __expf(v2 - mx) +
                  __expf(v3 - mx) + __expf(v4 - mx);
  const float lse = mx + __logf(s);
  p[0] = v0 - lse; p[1] = v1 - lse; p[2] = v2 - lse; p[3] = v3 - lse; p[4] = v4 - lse;
}

// ---------------------------------------------------------------------------
extern "C" void kernel_launch(void* const* d_in, const int* in_sizes, int n_in,
                              void* d_out, int out_size, void* d_ws, size_t ws_size,
                              hipStream_t stream) {
  (void)in_sizes; (void)n_in; (void)out_size; (void)ws_size;
  const float* z       = (const float*)d_in[0];
  const int*   actions = (const int*)d_in[1];
  const int*   dones   = (const int*)d_in[2];
  const float* embed   = (const float*)d_in[3];
  const float* W_ih    = (const float*)d_in[4];
  const float* W_hh    = (const float*)d_in[5];
  const float* b_ih    = (const float*)d_in[6];
  const float* b_hh    = (const float*)d_in[7];
  const float* W_out   = (const float*)d_in[8];
  const float* b_out   = (const float*)d_in[9];

  // workspace layout (bytes)
  char* ws = (char*)d_ws;
  size_t off = 0;
  float*  x_pre   = (float*)(ws + off);  off += (size_t)ROWS1 * G4 * 4;   // 268.4 MB
  __bf16* Wih_bf  = (__bf16*)(ws + off); off += (size_t)G4 * FP * 2;      // 1.3 MB
  __bf16* Whh_bf  = (__bf16*)(ws + off); off += (size_t)G4 * H_ * 2;      // 8.4 MB
  __bf16* Wout_bf = (__bf16*)(ws + off); off += (size_t)OUTP * H_ * 2;    // 4.0 MB
  float*  bias    = (float*)(ws + off);  off += (size_t)G4 * 4;
  __bf16* inp_bf  = (__bf16*)(ws + off); off += (size_t)ROWS1 * FP * 2;   // 5.2 MB
  __bf16* hs      = (__bf16*)(ws + off); off += (size_t)ROWS3 * H_ * 2;   // 33.5 MB
  __bf16* h0      = (__bf16*)(ws + off); off += (size_t)B_ * H_ * 2;      // 64 KB
  __bf16* h1      = (__bf16*)(ws + off); off += (size_t)B_ * H_ * 2;      // 64 KB
  unsigned int* ctr = (unsigned int*)(ws + off); off += 256;

  float* o_logmix = (float*)d_out;
  float* o_mu     = o_logmix + (size_t)ROWS3 * KMIX;
  float* o_logstd = o_mu + (size_t)ROWS3 * KMIX;
  float* o_done   = o_logstd + (size_t)ROWS3 * KMIX;

  prep_kernel<<<2048, 256, 0, stream>>>(z, actions, embed, W_ih, W_hh, b_ih, b_hh, W_out,
                                        inp_bf, Wih_bf, Whh_bf, Wout_bf, bias, h0, h1, ctr);

  xpre_kernel<<<dim3(G4 / 128, ROWS1 / 16), 256, 0, stream>>>(inp_bf, Wih_bf, bias, x_pre);

  lstm_kernel<<<NWG, 32 * LWAVES, 0, stream>>>(x_pre, dones, Whh_bf, h0, h1, hs, ctr);

  out_kernel<<<dim3((OUTP / 16 + 7) / 8, ROWS3 / 16), 256, 0, stream>>>(
      hs, Wout_bf, b_out, o_logmix, o_mu, o_logstd, o_done);

  const long ngroups = (long)ROWS3 * (KMIX / 5);
  lse_kernel<<<(ngroups + 255) / 256, 256, 0, stream>>>(o_logmix, ngroups);
}
